// TrajectoryPredictor_85229331022215
// MI455X (gfx1250) — matrix operand order repaired and checked by reading, not compiled, not run on hardware
//
#include <hip/hip_runtime.h>

#define HID   256
#define TLEN  10
#define PRED  30
#define MROWS 64      // batch rows per workgroup
#define HPAD  264     // f16 row stride (halves) for h arrays -> 4-bank shift/row
#define CPAD  260     // f32 row stride (floats) for c arrays

typedef _Float16 v8h  __attribute__((ext_vector_type(8)));
typedef _Float16 v16h __attribute__((ext_vector_type(16)));
typedef float    v8f  __attribute__((ext_vector_type(8)));

static __device__ __forceinline__ v16h cat16(v8h lo, v8h hi) {
  return __builtin_shufflevector(lo, hi, 0,1,2,3,4,5,6,7,8,9,10,11,12,13,14,15);
}
static __device__ __forceinline__ float sigmoid_(float x) {
  return 1.0f / (1.0f + __expf(-x));
}
static __device__ __forceinline__ float tanh_(float x) {
  float e = __expf(2.0f * x);          // stable: +inf -> 1, 0 -> -1
  return 1.0f - 2.0f / (e + 1.0f);
}

// Convert weights to f16; build concatenated layer-1 weight [Wih1 ; Whh1] (1024 x 512)
__global__ void prep_weights(const float* __restrict__ Whh0,
                             const float* __restrict__ Wih1,
                             const float* __restrict__ Whh1,
                             _Float16* __restrict__ W0h,   // 1024 x 256
                             _Float16* __restrict__ W1) {  // 1024 x 512
  int idx = blockIdx.x * blockDim.x + threadIdx.x;
  if (idx < 1024 * 256) W0h[idx] = (_Float16)Whh0[idx];
  if (idx < 1024 * 512) {
    int n = idx >> 9, k = idx & 511;
    float v = (k < 256) ? Wih1[n * 256 + k] : Whh1[n * 256 + (k - 256)];
    W1[idx] = (_Float16)v;
  }
}

__global__ __launch_bounds__(256)
void lstm_traj(const float* __restrict__ x,
               const float* __restrict__ Wih0,
               const float* __restrict__ bih0, const float* __restrict__ bhh0,
               const float* __restrict__ bih1, const float* __restrict__ bhh1,
               const float* __restrict__ Wdec, const float* __restrict__ bdec,
               const _Float16* __restrict__ W0h, const _Float16* __restrict__ W1,
               float* __restrict__ out) {
  extern __shared__ char smem[];
  _Float16* H0 = (_Float16*)smem;                 // MROWS x HPAD (f16)
  _Float16* H1 = H0 + MROWS * HPAD;
  float*    C0 = (float*)(H1 + MROWS * HPAD);     // MROWS x CPAD (f32)
  float*    C1 = C0 + MROWS * CPAD;
  float*    XS = C1 + MROWS * CPAD;               // MROWS x TLEN x 3

  const int tid  = threadIdx.x;
  const int lane = tid & 31;
  const int wv   = tid >> 5;      // wave 0..7 -> hidden cols [wv*32, wv*32+32)
  const int hi   = lane >> 4;     // half-wave
  const int ln   = lane & 15;
  const int batchBase = blockIdx.x * MROWS;

  // ---- init LDS state ----
  for (int i = tid; i < MROWS * HPAD; i += 256) { H0[i] = (_Float16)0.0f; H1[i] = (_Float16)0.0f; }
  for (int i = tid; i < MROWS * CPAD; i += 256) { C0[i] = 0.0f; C1[i] = 0.0f; }
  for (int i = tid; i < MROWS * TLEN * 3; i += 256) XS[i] = x[(size_t)batchBase * (TLEN * 3) + i];
  __syncthreads();

  // ---- per-lane per-tile constants (tile t8: gate = t8>>1, col-half = t8&1) ----
  float bias0[8], bias1[8], wx0[8], wx1[8], wx2[8];
  const _Float16* bp0[8];
  const _Float16* bp1[8];
#pragma unroll
  for (int t8 = 0; t8 < 8; ++t8) {
    int n = (t8 >> 1) * 256 + wv * 32 + (t8 & 1) * 16 + ln;   // gate output index
    bias0[t8] = bih0[n] + bhh0[n];
    bias1[t8] = bih1[n] + bhh1[n];
    wx0[t8] = Wih0[n * 3 + 0]; wx1[t8] = Wih0[n * 3 + 1]; wx2[t8] = Wih0[n * 3 + 2];
    bp0[t8] = W0h + n * 256 + hi * 8;    // B-frag: k = 8h+s (s<8), 16+8h+(s-8) (s>=8)
    bp1[t8] = W1  + n * 512 + hi * 8;
  }

  for (int t = 0; t < TLEN; ++t) {
    // ================= LAYER 0 : gates = x@Wih0^T + h0@Whh0^T + b =================
#pragma unroll 1
    for (int pair = 0; pair < 2; ++pair) {
      const int pb = pair * 32;
      v8f acc[2][8];
      // init with bias + tiny x (D=3) contribution
#pragma unroll
      for (int s = 0; s < 2; ++s) {
#pragma unroll
        for (int r = 0; r < 8; ++r) {
          int m = pb + s * 16 + r + 8 * hi;
          float x0 = XS[m * 30 + t * 3 + 0];
          float x1 = XS[m * 30 + t * 3 + 1];
          float x2 = XS[m * 30 + t * 3 + 2];
#pragma unroll
          for (int t8 = 0; t8 < 8; ++t8)
            acc[s][t8][r] = bias0[t8] + wx0[t8] * x0 + wx1[t8] * x1 + wx2[t8] * x2;
        }
      }
      // GEMM over K=256 (h0[t-1])
#pragma unroll 2
      for (int ks = 0; ks < 8; ++ks) {
        int kh = ks * 32 + hi * 8;
        const _Float16* a0p = H0 + (pb + ln) * HPAD + kh;
        const _Float16* a1p = H0 + (pb + 16 + ln) * HPAD + kh;
        v16h a0 = cat16(*(const v8h*)a0p, *(const v8h*)(a0p + 16));
        v16h a1 = cat16(*(const v8h*)a1p, *(const v8h*)(a1p + 16));
#pragma unroll
        for (int t8 = 0; t8 < 8; ++t8) {
          const _Float16* bp = bp0[t8] + ks * 32;
          v16h b = cat16(*(const v8h*)bp, *(const v8h*)(bp + 16));
          acc[0][t8] = __builtin_amdgcn_wmma_f32_16x16x32_f16(false, a0, false, b, (short)0, acc[0][t8], false, false);
          acc[1][t8] = __builtin_amdgcn_wmma_f32_16x16x32_f16(false, a1, false, b, (short)0, acc[1][t8], false, false);
        }
      }
      __syncthreads();   // all reads of H0 (t-1) done before updates
      // element-wise gate update -> C0, H0
#pragma unroll
      for (int s = 0; s < 2; ++s) {
#pragma unroll
        for (int col = 0; col < 2; ++col) {
          int j = wv * 32 + col * 16 + ln;
#pragma unroll
          for (int r = 0; r < 8; ++r) {
            int m = pb + s * 16 + r + 8 * hi;
            float iv = sigmoid_(acc[s][0 + col][r]);
            float fv = sigmoid_(acc[s][2 + col][r]);
            float gv = tanh_   (acc[s][4 + col][r]);
            float ov = sigmoid_(acc[s][6 + col][r]);
            float c  = fv * C0[m * CPAD + j] + iv * gv;
            C0[m * CPAD + j] = c;
            H0[m * HPAD + j] = (_Float16)(ov * tanh_(c));
          }
        }
      }
      __syncthreads();
    }

    // ======= LAYER 1 : gates = [h0(t), h1(t-1)] (K=512) @ [Wih1;Whh1]^T + b =======
#pragma unroll 1
    for (int pair = 0; pair < 2; ++pair) {
      const int pb = pair * 32;
      v8f acc[2][8];
#pragma unroll
      for (int s = 0; s < 2; ++s)
#pragma unroll
        for (int t8 = 0; t8 < 8; ++t8)
#pragma unroll
          for (int r = 0; r < 8; ++r) acc[s][t8][r] = bias1[t8];
#pragma unroll 2
      for (int ks = 0; ks < 16; ++ks) {
        const _Float16* hsrc = (ks < 8) ? H0 : H1;
        int kh = (ks & 7) * 32 + hi * 8;
        const _Float16* a0p = hsrc + (pb + ln) * HPAD + kh;
        const _Float16* a1p = hsrc + (pb + 16 + ln) * HPAD + kh;
        v16h a0 = cat16(*(const v8h*)a0p, *(const v8h*)(a0p + 16));
        v16h a1 = cat16(*(const v8h*)a1p, *(const v8h*)(a1p + 16));
#pragma unroll
        for (int t8 = 0; t8 < 8; ++t8) {
          const _Float16* bp = bp1[t8] + ks * 32;
          v16h b = cat16(*(const v8h*)bp, *(const v8h*)(bp + 16));
          acc[0][t8] = __builtin_amdgcn_wmma_f32_16x16x32_f16(false, a0, false, b, (short)0, acc[0][t8], false, false);
          acc[1][t8] = __builtin_amdgcn_wmma_f32_16x16x32_f16(false, a1, false, b, (short)0, acc[1][t8], false, false);
        }
      }
      __syncthreads();
#pragma unroll
      for (int s = 0; s < 2; ++s) {
#pragma unroll
        for (int col = 0; col < 2; ++col) {
          int j = wv * 32 + col * 16 + ln;
#pragma unroll
          for (int r = 0; r < 8; ++r) {
            int m = pb + s * 16 + r + 8 * hi;
            float iv = sigmoid_(acc[s][0 + col][r]);
            float fv = sigmoid_(acc[s][2 + col][r]);
            float gv = tanh_   (acc[s][4 + col][r]);
            float ov = sigmoid_(acc[s][6 + col][r]);
            float c  = fv * C1[m * CPAD + j] + iv * gv;
            C1[m * CPAD + j] = c;
            H1[m * HPAD + j] = (_Float16)(ov * tanh_(c));
          }
        }
      }
      __syncthreads();
    }
  }

  // ---- decoder: velocity = h1_T @ Wdec^T + bdec ; rollout 30 steps ----
  if (tid < MROWS * 3) {
    int m = tid / 3, d = tid - 3 * (tid / 3);
    float v = bdec[d];
    const float* wd = Wdec + d * 256;
#pragma unroll 8
    for (int j = 0; j < 256; ++j) v += (float)H1[m * HPAD + j] * wd[j];
    float pos = XS[m * 30 + 27 + d];                 // last position (t=9)
    size_t ob = ((size_t)(batchBase + m)) * (PRED * 3) + d;
#pragma unroll
    for (int s = 0; s < PRED; ++s) out[ob + (size_t)s * 3] = pos + v * (float)(s + 1);
  }
}

extern "C" void kernel_launch(void* const* d_in, const int* in_sizes, int n_in,
                              void* d_out, int out_size, void* d_ws, size_t ws_size,
                              hipStream_t stream) {
  const float* x    = (const float*)d_in[0];
  const float* Wih0 = (const float*)d_in[1];
  const float* Whh0 = (const float*)d_in[2];
  const float* bih0 = (const float*)d_in[3];
  const float* bhh0 = (const float*)d_in[4];
  const float* Wih1 = (const float*)d_in[5];
  const float* Whh1 = (const float*)d_in[6];
  const float* bih1 = (const float*)d_in[7];
  const float* bhh1 = (const float*)d_in[8];
  const float* Wdec = (const float*)d_in[9];
  const float* bdec = (const float*)d_in[10];
  float* out = (float*)d_out;

  _Float16* W0h = (_Float16*)d_ws;          // 1024*256 f16
  _Float16* W1  = W0h + 1024 * 256;         // 1024*512 f16  (total ~1.5 MB of ws)

  int B = in_sizes[0] / (TLEN * 3);

  prep_weights<<<(1024 * 512 + 255) / 256, 256, 0, stream>>>(Whh0, Wih1, Whh1, W0h, W1);

  size_t smemBytes = (size_t)(2 * MROWS * HPAD) * sizeof(_Float16)
                   + (size_t)(2 * MROWS * CPAD) * sizeof(float)
                   + (size_t)(MROWS * TLEN * 3) * sizeof(float);   // ~203 KB (<320 KB WGP LDS)

  lstm_traj<<<B / MROWS, 256, smemBytes, stream>>>(x, Wih0, bih0, bhh0, bih1, bhh1,
                                                   Wdec, bdec, W0h, W1, out);
}
